// UnfedAttention_88390426952113
// MI455X (gfx1250) — matrix operand order repaired and checked by reading, not compiled, not run on hardware
//
#include <hip/hip_runtime.h>
#include <hip/hip_bf16.h>

// ---------------------------------------------------------------------------
// Problem constants (from reference)
// ---------------------------------------------------------------------------
#define BB   2
#define SS   2048
#define HH   2048
#define NHQ  16
#define NKV  4
#define GG   (NHQ / NKV)     // 4
#define HD   128
#define TOK  (BB * SS)       // 4096

typedef __bf16 bf16_t;
typedef __attribute__((ext_vector_type(16))) __bf16 bf16x16;
typedef __attribute__((ext_vector_type(8)))  __bf16 bf16x8;
typedef __attribute__((ext_vector_type(4)))  __bf16 bf16x4;
typedef __attribute__((ext_vector_type(8)))  float  v8f;

union ABFrag {
    bf16x16 v;
    bf16x8  h[2];
};

__device__ __forceinline__ v8f wmma_bf16(bf16x16 a, bf16x16 b, v8f c) {
    // (neg_a, A, neg_b, B, c_mod, C, reuse_a, reuse_b)
    return __builtin_amdgcn_wmma_f32_16x16x32_bf16(false, a, false, b, (short)0, c,
                                                   false, false);
}

// ---------------------------------------------------------------------------
// 1) f32 -> bf16 convert (hidden states)
// ---------------------------------------------------------------------------
__global__ __launch_bounds__(256) void cvt_bf16_kernel(const float* __restrict__ src,
                                                       bf16_t* __restrict__ dst, int n4) {
    int i = blockIdx.x * blockDim.x + threadIdx.x;
    if (i >= n4) return;
    float4 v = reinterpret_cast<const float4*>(src)[i];
    bf16x4 o = { (__bf16)v.x, (__bf16)v.y, (__bf16)v.z, (__bf16)v.w };
    reinterpret_cast<bf16x4*>(dst)[i] = o;
}

// ---------------------------------------------------------------------------
// 2) Transpose + convert: src f32 [K][N]  ->  dst bf16 [N][K]
// ---------------------------------------------------------------------------
__global__ __launch_bounds__(256) void transpose_cvt_kernel(const float* __restrict__ src,
                                                            bf16_t* __restrict__ dst,
                                                            int K, int N) {
    __shared__ float tile[32][33];
    int bx = blockIdx.x * 32;   // N offset
    int by = blockIdx.y * 32;   // K offset
    int tx = threadIdx.x & 31;
    int ty = threadIdx.x >> 5;  // 0..7
#pragma unroll
    for (int i = 0; i < 32; i += 8)
        tile[ty + i][tx] = src[(size_t)(by + ty + i) * N + bx + tx];
    __syncthreads();
#pragma unroll
    for (int i = 0; i < 32; i += 8)
        dst[(size_t)(bx + ty + i) * K + by + tx] = (bf16_t)tile[tx][ty + i];
}

// ---------------------------------------------------------------------------
// 3) WMMA GEMM:  C[M][N] = A[M][K] * Bt[N][K]^T     (bf16 in, f32 acc)
//    K is fixed at 2048 for every GEMM in this pipeline -> software pipeline
//    with ping-pong operand buffers so loads overlap WMMAs.
//    block = 128 threads (4 waves); wave w -> rows [blockIdx.y*64 + w*16, +16)
//    block covers 64 columns (4 n-tiles of 16).
// ---------------------------------------------------------------------------
__global__ __launch_bounds__(128) void gemm_bf16_kernel(const bf16_t* __restrict__ A,
                                                        const bf16_t* __restrict__ Bt,
                                                        void* __restrict__ Cout,
                                                        int M, int N, int out_bf16) {
    constexpr int KD = 2048;

    int w    = threadIdx.x >> 5;
    int lane = threadIdx.x & 31;
    int half = lane >> 4;
    int r    = lane & 15;

    int mbase = (blockIdx.y * 4 + w) * 16;
    int nbase = blockIdx.x * 64;

    const bf16_t* arow = A + (size_t)(mbase + r) * KD;
    const bf16_t* bp[4];
#pragma unroll
    for (int nt = 0; nt < 4; ++nt)
        bp[nt] = Bt + (size_t)(nbase + nt * 16 + r) * KD + half * 8;

    ABFrag a[2];
    ABFrag b[2][4];

    auto loadAB = [&](int buf, int k0) {
        a[buf].h[0] = *reinterpret_cast<const bf16x8*>(arow + k0 + half * 8);
        a[buf].h[1] = *reinterpret_cast<const bf16x8*>(arow + k0 + 16 + half * 8);
#pragma unroll
        for (int nt = 0; nt < 4; ++nt) {
            b[buf][nt].h[0] = *reinterpret_cast<const bf16x8*>(bp[nt] + k0);
            b[buf][nt].h[1] = *reinterpret_cast<const bf16x8*>(bp[nt] + k0 + 16);
        }
    };

    v8f acc[4];
#pragma unroll
    for (int nt = 0; nt < 4; ++nt) acc[nt] = (v8f)0.0f;

    auto compute = [&](int buf) {
#pragma unroll
        for (int nt = 0; nt < 4; ++nt)
            acc[nt] = wmma_bf16(a[buf].v, b[buf][nt].v, acc[nt]);
    };

    // prologue
    loadAB(0, 0);
    // steady state: while computing buf X, loads for buf !X are in flight
#pragma unroll 1
    for (int k0 = 0; k0 < KD - 64; k0 += 64) {
        loadAB(1, k0 + 32);
        compute(0);
        loadAB(0, k0 + 64);
        compute(1);
    }
    // epilogue
    loadAB(1, KD - 32);
    compute(0);
    compute(1);

#pragma unroll
    for (int nt = 0; nt < 4; ++nt) {
#pragma unroll
        for (int j = 0; j < 8; ++j) {
            int row = mbase + j + half * 8;
            int col = nbase + nt * 16 + r;
            if (out_bf16)
                ((bf16_t*)Cout)[(size_t)row * N + col] = (bf16_t)acc[nt][j];
            else
                ((float*)Cout)[(size_t)row * N + col] = acc[nt][j];
        }
    }
}

// ---------------------------------------------------------------------------
// 4) RoPE + repack:  raw bf16 [TOK][nh*128] -> dst bf16 [B][nh][S][HD]
// ---------------------------------------------------------------------------
__global__ __launch_bounds__(256) void rope_kernel(const bf16_t* __restrict__ raw,
                                                   const float* __restrict__ cosp,
                                                   const float* __restrict__ sinp,
                                                   bf16_t* __restrict__ dst,
                                                   int nh, int total) {
    int idx = blockIdx.x * blockDim.x + threadIdx.x;
    if (idx >= total) return;
    int d   = idx & 63;
    int t2  = idx >> 6;
    int h   = t2 % nh;
    int tok = t2 / nh;
    int b   = tok >> 11;     // / SS
    int s   = tok & 2047;    // % SS

    size_t rbase = (size_t)tok * ((size_t)nh * HD) + (size_t)h * HD;
    float x1 = (float)raw[rbase + d];
    float x2 = (float)raw[rbase + d + 64];

    size_t cbase = (size_t)tok * HD;
    float c1 = cosp[cbase + d],      s1 = sinp[cbase + d];
    float c2 = cosp[cbase + d + 64], s2 = sinp[cbase + d + 64];

    size_t obase = (((size_t)b * nh + h) * SS + s) * HD;
    dst[obase + d]      = (bf16_t)(x1 * c1 - x2 * s1);
    dst[obase + d + 64] = (bf16_t)(x2 * c2 + x1 * s2);
}

// ---------------------------------------------------------------------------
// 5) V transpose: raw bf16 [TOK][NKV*HD] -> Vt bf16 [B][NKV][HD][S]
// ---------------------------------------------------------------------------
__global__ __launch_bounds__(256) void vtrans_kernel(const bf16_t* __restrict__ raw,
                                                     bf16_t* __restrict__ dst) {
    __shared__ bf16_t tile[32][33];
    int bk  = blockIdx.z;
    int b   = bk / NKV;
    int kvh = bk % NKV;
    int s0  = blockIdx.x * 32;
    int d0  = blockIdx.y * 32;
    int tx  = threadIdx.x & 31;
    int ty  = threadIdx.x >> 5;
#pragma unroll
    for (int i = 0; i < 32; i += 8)
        tile[ty + i][tx] =
            raw[(size_t)(b * SS + s0 + ty + i) * (NKV * HD) + kvh * HD + d0 + tx];
    __syncthreads();
#pragma unroll
    for (int i = 0; i < 32; i += 8)
        dst[(((size_t)b * NKV + kvh) * HD + d0 + ty + i) * SS + s0 + tx] =
            tile[tx][ty + i];
}

// ---------------------------------------------------------------------------
// 6) Flash attention (causal, GQA): Q[B][NH][S][HD], K[B][NKV][S][HD],
//    Vt[B][NKV][HD][S] -> AO bf16 [TOK][NH*HD]
//    4 waves/block, each wave owns 16 q rows; kv tiles of 64; online softmax.
//    Next kv tile's K/V rows are prefetched (global_prefetch_b8) while the
//    current tile computes.
// ---------------------------------------------------------------------------
__global__ __launch_bounds__(128) void flash_kernel(const bf16_t* __restrict__ Q,
                                                    const bf16_t* __restrict__ K,
                                                    const bf16_t* __restrict__ Vt,
                                                    bf16_t* __restrict__ AO) {
    __shared__ __align__(16) bf16_t lds[4][16][72];   // per-wave 16x64 P tile

    int w    = threadIdx.x >> 5;
    int lane = threadIdx.x & 31;
    int half = lane >> 4;
    int r    = lane & 15;

    int qb  = blockIdx.x * 64;      // block q base
    int h   = blockIdx.y;           // query head
    int b   = blockIdx.z;
    int kvh = h / GG;

    const bf16_t* Qp = Q  + (((size_t)b * NHQ + h)   * SS) * HD;
    const bf16_t* Kp = K  + (((size_t)b * NKV + kvh) * SS) * HD;
    const bf16_t* Vp = Vt + (((size_t)b * NKV + kvh) * HD) * SS;

    int wq   = qb + w * 16;         // this wave's q base
    int qrow = wq + r;

    // Q A-operands: 4 k-steps of 32 over HD=128
    ABFrag qa[4];
#pragma unroll
    for (int kk = 0; kk < 4; ++kk) {
        const bf16_t* qptr = Qp + (size_t)qrow * HD + kk * 32;
        qa[kk].h[0] = *reinterpret_cast<const bf16x8*>(qptr + half * 8);
        qa[kk].h[1] = *reinterpret_cast<const bf16x8*>(qptr + 16 + half * 8);
    }

    v8f o[8];
#pragma unroll
    for (int dt = 0; dt < 8; ++dt) o[dt] = (v8f)0.0f;
    float mrow[8], lrow[8];
#pragma unroll
    for (int j = 0; j < 8; ++j) { mrow[j] = -3.0e38f; lrow[j] = 0.0f; }

    const float scale = 0.08838834764831845f;   // 1/sqrt(128)
    int nsteps = blockIdx.x + 1;

    for (int t = 0; t < nsteps; ++t) {
        int kb = t * 64;

        // ---- prefetch next kv tile (wave-uniform condition) ----
        if (t + 1 < nsteps) {
            int kn = kb + 64;
#pragma unroll
            for (int nt = 0; nt < 4; ++nt)
                __builtin_prefetch(Kp + (size_t)(kn + nt * 16 + r) * HD, 0, 1);
#pragma unroll
            for (int dt = 0; dt < 8; ++dt)
                __builtin_prefetch(Vp + (size_t)(dt * 16 + r) * SS + kn, 0, 1);
        }

        // ---- scores: S = Q @ K^T (16 x 64) ----
        v8f sacc[4];
#pragma unroll
        for (int nt = 0; nt < 4; ++nt) sacc[nt] = (v8f)0.0f;
#pragma unroll
        for (int nt = 0; nt < 4; ++nt) {
            const bf16_t* kp = Kp + (size_t)(kb + nt * 16 + r) * HD;
            v8f a = sacc[nt];
#pragma unroll
            for (int kk = 0; kk < 4; ++kk) {
                ABFrag bfr;
                bfr.h[0] = *reinterpret_cast<const bf16x8*>(kp + kk * 32 + half * 8);
                bfr.h[1] = *reinterpret_cast<const bf16x8*>(kp + kk * 32 + 16 + half * 8);
                a = wmma_bf16(qa[kk].v, bfr.v, a);
            }
            sacc[nt] = a;
        }

        // scale + causal mask (only the diagonal kv tile needs masking)
#pragma unroll
        for (int nt = 0; nt < 4; ++nt)
#pragma unroll
            for (int j = 0; j < 8; ++j) sacc[nt][j] *= scale;

        if (t == nsteps - 1) {
#pragma unroll
            for (int nt = 0; nt < 4; ++nt) {
                int kc = kb + nt * 16 + r;
#pragma unroll
                for (int j = 0; j < 8; ++j) {
                    int qr = wq + j + half * 8;
                    sacc[nt][j] = (kc > qr) ? -1.0e30f : sacc[nt][j];
                }
            }
        }

        // ---- online softmax (row stats across 16-lane halves) ----
#pragma unroll
        for (int j = 0; j < 8; ++j) {
            float mx = fmaxf(fmaxf(sacc[0][j], sacc[1][j]),
                             fmaxf(sacc[2][j], sacc[3][j]));
            mx = fmaxf(mx, __shfl_xor(mx, 1));
            mx = fmaxf(mx, __shfl_xor(mx, 2));
            mx = fmaxf(mx, __shfl_xor(mx, 4));
            mx = fmaxf(mx, __shfl_xor(mx, 8));
            float mnew  = fmaxf(mrow[j], mx);
            float alpha = __expf(mrow[j] - mnew);
            mrow[j] = mnew;
            float rs = 0.0f;
#pragma unroll
            for (int nt = 0; nt < 4; ++nt) {
                float p = __expf(sacc[nt][j] - mnew);
                sacc[nt][j] = p;
                rs += p;
            }
            rs += __shfl_xor(rs, 1);
            rs += __shfl_xor(rs, 2);
            rs += __shfl_xor(rs, 4);
            rs += __shfl_xor(rs, 8);
            lrow[j] = lrow[j] * alpha + rs;
#pragma unroll
            for (int dt = 0; dt < 8; ++dt) o[dt][j] *= alpha;
        }

        // ---- P (C-layout) -> LDS -> A-operand layout ----
#pragma unroll
        for (int nt = 0; nt < 4; ++nt)
#pragma unroll
            for (int j = 0; j < 8; ++j)
                lds[w][j + half * 8][nt * 16 + r] = (bf16_t)sacc[nt][j];

        asm volatile("s_wait_dscnt 0" ::: "memory");

        ABFrag pa[2];
#pragma unroll
        for (int kk = 0; kk < 2; ++kk) {
            pa[kk].h[0] = *reinterpret_cast<const bf16x8*>(
                &lds[w][r][kk * 32 + half * 8]);
            pa[kk].h[1] = *reinterpret_cast<const bf16x8*>(
                &lds[w][r][kk * 32 + 16 + half * 8]);
        }

        // ---- O += P @ V  (16 x 128) ----
#pragma unroll
        for (int dt = 0; dt < 8; ++dt) {
            const bf16_t* vp = Vp + (size_t)(dt * 16 + r) * SS + kb;
            v8f a = o[dt];
#pragma unroll
            for (int kk = 0; kk < 2; ++kk) {
                ABFrag bfr;
                bfr.h[0] = *reinterpret_cast<const bf16x8*>(vp + kk * 32 + half * 8);
                bfr.h[1] = *reinterpret_cast<const bf16x8*>(vp + kk * 32 + 16 + half * 8);
                a = wmma_bf16(pa[kk].v, bfr.v, a);
            }
            o[dt] = a;
        }
    }

    // ---- epilogue: normalize, store AO[tok][h*128 + d] ----
#pragma unroll
    for (int j = 0; j < 8; ++j) {
        float inv = 1.0f / lrow[j];
        int row   = wq + j + half * 8;
        size_t base = ((size_t)b * SS + row) * (NHQ * HD) + (size_t)h * HD;
#pragma unroll
        for (int dt = 0; dt < 8; ++dt)
            AO[base + dt * 16 + r] = (bf16_t)(o[dt][j] * inv);
    }
}

// ---------------------------------------------------------------------------
// Host orchestration
// ---------------------------------------------------------------------------
extern "C" void kernel_launch(void* const* d_in, const int* in_sizes, int n_in,
                              void* d_out, int out_size, void* d_ws, size_t ws_size,
                              hipStream_t stream) {
    const float* hs   = (const float*)d_in[0];
    const float* cosp = (const float*)d_in[1];
    const float* sinp = (const float*)d_in[2];
    const float* Wq   = (const float*)d_in[3];
    const float* Wk   = (const float*)d_in[4];
    const float* Wv   = (const float*)d_in[5];
    const float* Wo   = (const float*)d_in[6];

    char* ws = (char*)d_ws;
    size_t off = 0;
    auto alloc = [&](size_t bytes) -> void* {
        void* p = ws + off;
        off += (bytes + 255) & ~(size_t)255;
        return p;
    };

    bf16_t* HS   = (bf16_t*)alloc((size_t)TOK * HH * 2);          // 16 MB
    bf16_t* WQT  = (bf16_t*)alloc((size_t)2048 * 2048 * 2);       //  8 MB
    bf16_t* WKT  = (bf16_t*)alloc((size_t)512  * 2048 * 2);       //  2 MB
    bf16_t* WVT  = (bf16_t*)alloc((size_t)512  * 2048 * 2);       //  2 MB
    bf16_t* WOT  = (bf16_t*)alloc((size_t)2048 * 2048 * 2);       //  8 MB
    bf16_t* QR   = (bf16_t*)alloc((size_t)BB * NHQ * SS * HD * 2); // 16 MB
    bf16_t* KR   = (bf16_t*)alloc((size_t)BB * NKV * SS * HD * 2); //  4 MB
    bf16_t* VT   = (bf16_t*)alloc((size_t)BB * NKV * HD * SS * 2); //  4 MB
    bf16_t* QRAW = (bf16_t*)alloc((size_t)TOK * 2048 * 2);        // 16 MB
    bf16_t* KRAW = (bf16_t*)alloc((size_t)TOK * 512 * 2);         //  4 MB
    bf16_t* VRAW = (bf16_t*)alloc((size_t)TOK * 512 * 2);         //  4 MB
    bf16_t* AO   = QRAW;  // reuse: QRAW is dead once rope_kernel(Q) finished

    // 1) hidden -> bf16
    {
        int n4 = (TOK * HH) / 4;
        cvt_bf16_kernel<<<n4 / 256, 256, 0, stream>>>(hs, HS, n4);
    }
    // 2) weight transposes (f32 [K][N] -> bf16 [N][K])
    transpose_cvt_kernel<<<dim3(2048 / 32, 2048 / 32), 256, 0, stream>>>(Wq, WQT, 2048, 2048);
    transpose_cvt_kernel<<<dim3(512  / 32, 2048 / 32), 256, 0, stream>>>(Wk, WKT, 2048, 512);
    transpose_cvt_kernel<<<dim3(512  / 32, 2048 / 32), 256, 0, stream>>>(Wv, WVT, 2048, 512);
    transpose_cvt_kernel<<<dim3(2048 / 32, 2048 / 32), 256, 0, stream>>>(Wo, WOT, 2048, 2048);

    // 3) QKV projections (bf16 out)
    gemm_bf16_kernel<<<dim3(2048 / 64, TOK / 64), 128, 0, stream>>>(HS, WQT, QRAW, TOK, 2048, 1);
    gemm_bf16_kernel<<<dim3(512  / 64, TOK / 64), 128, 0, stream>>>(HS, WKT, KRAW, TOK, 512, 1);
    gemm_bf16_kernel<<<dim3(512  / 64, TOK / 64), 128, 0, stream>>>(HS, WVT, VRAW, TOK, 512, 1);

    // 4) RoPE + repack
    {
        int totq = TOK * NHQ * 64;
        rope_kernel<<<totq / 256, 256, 0, stream>>>(QRAW, cosp, sinp, QR, NHQ, totq);
        int totk = TOK * NKV * 64;
        rope_kernel<<<totk / 256, 256, 0, stream>>>(KRAW, cosp, sinp, KR, NKV, totk);
    }
    // 5) V transpose
    vtrans_kernel<<<dim3(SS / 32, HD / 32, BB * NKV), 256, 0, stream>>>(VRAW, VT);

    // 6) flash attention
    flash_kernel<<<dim3(SS / 64, NHQ, BB), 128, 0, stream>>>(QR, KR, VT, AO);

    // 7) output projection (f32 out)
    gemm_bf16_kernel<<<dim3(2048 / 64, TOK / 64), 128, 0, stream>>>(AO, WOT, d_out, TOK, 2048, 0);

    (void)in_sizes; (void)n_in; (void)out_size; (void)ws_size;
}
